// GCN_70325794504771
// MI455X (gfx1250) — compile-verified, matrix operand order checked by reading, not run on hardware
//
#include <hip/hip_runtime.h>
#include <hip/hip_bf16.h>

#define NN 10000
#define EE 640000
#define DD 128
#define CC 10
#define GG 64

typedef __attribute__((ext_vector_type(16))) _Float16 v16h;
typedef __attribute__((ext_vector_type(8)))  _Float16 v8h;
typedef __attribute__((ext_vector_type(8)))  float    v8f;

// ---------------- degree kernels ----------------
__global__ void k_deg_init(float* __restrict__ deg) {
    int i = blockIdx.x * blockDim.x + threadIdx.x;
    if (i < NN) deg[i] = 1.0f;   // self-loop contributes 1
}

__global__ void k_deg_count(const long long* __restrict__ dst, float* __restrict__ deg) {
    int e = blockIdx.x * blockDim.x + threadIdx.x;
    if (e < EE) atomicAdd(&deg[(int)dst[e]], 1.0f);
}

__global__ void k_deg_rsqrt(float* __restrict__ deg) {
    int i = blockIdx.x * blockDim.x + threadIdx.x;
    if (i < NN) deg[i] = rsqrtf(deg[i]);
}

// ---------------- weight prep: Wt[l][n*DD+k] = (f16) W_l[k*DD+n] ----------------
__global__ void k_wprep(const float* __restrict__ W0, const float* __restrict__ W1,
                        const float* __restrict__ W2, const float* __restrict__ W3,
                        _Float16* __restrict__ Wt) {
    int idx = blockIdx.x * blockDim.x + threadIdx.x;       // 4*DD*DD
    if (idx >= 4 * DD * DD) return;
    int l = idx >> 14;               // /16384
    int r = idx & (DD * DD - 1);
    int n = r >> 7;                  // /128
    int k = r & (DD - 1);
    const float* W = (l == 0) ? W0 : (l == 1) ? W1 : (l == 2) ? W2 : W3;
    Wt[(size_t)l * DD * DD + (size_t)n * DD + k] = (_Float16)W[(size_t)k * DD + n];
}

// ---------------- f32 -> f16 conversion (input layer only) ----------------
__global__ void k_tohalf(const float* __restrict__ x, _Float16* __restrict__ xh, int n) {
    int i = blockIdx.x * blockDim.x + threadIdx.x;
    if (i < n) xh[i] = (_Float16)x[i];
}

// ---------------- WMMA GEMM + fused self-loop init ----------------
// H[N x 128] = Ah[N x 128] * W  (Bt = W^T in f16);  agg = H * dis[row]^2
// grid.x = NN/16 = 625 row tiles; block = 256 threads = 8 waves, wave w -> column tile w.
__global__ void __launch_bounds__(256) k_gemm_wmma(const _Float16* __restrict__ Ah,
                                                   const _Float16* __restrict__ Bt,
                                                   const float* __restrict__ dis,
                                                   float* __restrict__ H,
                                                   float* __restrict__ agg) {
    const int lane = threadIdx.x & 31;
    const int wave = threadIdx.x >> 5;       // 0..7 -> column tile
    const int half = lane >> 4;              // 0 or 1
    const int sub  = lane & 15;

    const int r0 = blockIdx.x * 16;          // row tile base
    const int n0 = wave * 16;                // col tile base

    const int arow = r0 + sub;               // A row owned by this lane
    const int bcol = n0 + sub;               // B column owned by this lane

    v8f acc = {};
#pragma unroll
    for (int k0 = 0; k0 < DD; k0 += 32) {
        // A fragment (16x32 f16): lanes 0-15 K in [k0, k0+8)+[k0+16, k0+24),
        //                         lanes 16-31 K in [k0+8,k0+16)+[k0+24,k0+32)
        const _Float16* ap = Ah + (size_t)arow * DD + k0 + half * 8;
        v8h alo = *(const v8h*)(ap);
        v8h ahi = *(const v8h*)(ap + 16);
        // B fragment (32x16 f16), Bt is column-major of W so contiguous in K
        const _Float16* bp = Bt + (size_t)bcol * DD + k0 + half * 8;
        v8h blo = *(const v8h*)(bp);
        v8h bhi = *(const v8h*)(bp + 16);

        v16h a, b;
#pragma unroll
        for (int j = 0; j < 8; ++j) {
            a[j] = alo[j];  a[8 + j] = ahi[j];
            b[j] = blo[j];  b[8 + j] = bhi[j];
        }
        acc = __builtin_amdgcn_wmma_f32_16x16x32_f16(
            /*neg_a=*/false, a, /*neg_b=*/false, b,
            /*c_mod=*/(short)0, acc, /*reuse_a=*/false, /*reuse_b=*/false);
    }

    // D layout: lane -> col n0+sub ; VGPR v -> row r0 + v + 8*half
    // Fused epilogue: H = acc ; agg = acc * dis[row]^2 (self-loop term)
#pragma unroll
    for (int v = 0; v < 8; ++v) {
        int row = r0 + v + 8 * half;
        float d = dis[row];
        size_t o = (size_t)row * DD + n0 + sub;
        H[o]   = acc[v];
        agg[o] = acc[v] * d * d;
    }
}

// ---------------- edge scatter: agg[dst] += h[src]*norm ----------------
// one thread per (edge, 4-feature group): EE * 32 threads
__global__ void k_scatter(const long long* __restrict__ src, const long long* __restrict__ dst,
                          const float* __restrict__ dis, const float* __restrict__ h,
                          float* __restrict__ agg) {
    long long idx = (long long)blockIdx.x * blockDim.x + threadIdx.x;
    int e = (int)(idx >> 5);
    if (e >= EE) return;
    int g = (int)(idx & 31);
    int s = (int)src[e];
    int d = (int)dst[e];
    float norm = dis[s] * dis[d];
    const float4 v = *((const float4*)(h + (size_t)s * DD) + g);
    float* ap = agg + (size_t)d * DD + (size_t)g * 4;
    atomicAdd(ap + 0, v.x * norm);
    atomicAdd(ap + 1, v.y * norm);
    atomicAdd(ap + 2, v.z * norm);
    atomicAdd(ap + 3, v.w * norm);
}

// ---------------- fused ReLU + f32->f16 (feeds next layer's WMMA A) ----------------
__global__ void k_relu_tohalf(const float* __restrict__ a, float* __restrict__ o,
                              _Float16* __restrict__ oh, int n) {
    int i = blockIdx.x * blockDim.x + threadIdx.x;
    if (i < n) {
        float v = fmaxf(a[i], 0.0f);
        o[i]  = v;
        oh[i] = (_Float16)v;
    }
}

// ---------------- pooling ----------------
__global__ void k_zero(float* __restrict__ p, int n) {
    int i = blockIdx.x * blockDim.x + threadIdx.x;
    if (i < n) p[i] = 0.0f;
}

__global__ void k_pool(const float* __restrict__ x, const long long* __restrict__ batch,
                       float* __restrict__ xr) {
    int i = blockIdx.x * blockDim.x + threadIdx.x;
    if (i >= NN * DD) return;
    int n = i >> 7;
    int d = i & (DD - 1);
    atomicAdd(&xr[(size_t)((int)batch[n]) * DD + d], x[i]);
}

// ---------------- classifier + log_softmax ----------------
__global__ void k_classify(const float* __restrict__ x, const float* __restrict__ fcw,
                           const float* __restrict__ fcb, float* __restrict__ out) {
    int n = blockIdx.x * blockDim.x + threadIdx.x;
    if (n >= NN) return;
    float acc[CC];
#pragma unroll
    for (int c = 0; c < CC; ++c) acc[c] = fcb[c];
    const float* xr = x + (size_t)n * DD;
    for (int k = 0; k < DD; ++k) {
        float xv = xr[k];
        const float* wr = fcw + (size_t)k * CC;
#pragma unroll
        for (int c = 0; c < CC; ++c) acc[c] = fmaf(xv, wr[c], acc[c]);
    }
    float m = acc[0];
#pragma unroll
    for (int c = 1; c < CC; ++c) m = fmaxf(m, acc[c]);
    float s = 0.0f;
#pragma unroll
    for (int c = 0; c < CC; ++c) s += __expf(acc[c] - m);
    float lse = m + __logf(s);
    float* o = out + (size_t)n * CC;
#pragma unroll
    for (int c = 0; c < CC; ++c) o[c] = acc[c] - lse;
}

// ---------------- host-side launcher ----------------
static inline size_t align256(size_t x) { return (x + 255) & ~(size_t)255; }

extern "C" void kernel_launch(void* const* d_in, const int* in_sizes, int n_in,
                              void* d_out, int out_size, void* d_ws, size_t ws_size,
                              hipStream_t stream) {
    (void)in_sizes; (void)n_in; (void)out_size; (void)ws_size;

    const float*     x    = (const float*)d_in[0];
    const long long* ei   = (const long long*)d_in[1];   // [2, E] int64
    const long long* bat  = (const long long*)d_in[2];   // [N] int64
    const float*     W1   = (const float*)d_in[3];
    const float*     W2   = (const float*)d_in[4];
    const float*     W3   = (const float*)d_in[5];
    const float*     W4   = (const float*)d_in[6];
    const float*     fcw  = (const float*)d_in[7];
    const float*     fcb  = (const float*)d_in[8];

    const long long* src = ei;
    const long long* dst = ei + EE;

    float* out    = (float*)d_out;          // [N*C logits][G*D xr]
    float* xr_out = out + (size_t)NN * CC;

    // workspace layout (256B aligned regions)
    char* ws = (char*)d_ws;
    size_t off = 0;
    float*     dis  = (float*)(ws + off);     off = align256(off + sizeof(float) * NN);
    float*     h    = (float*)(ws + off);     off = align256(off + sizeof(float) * NN * DD);
    float*     agg  = (float*)(ws + off);     off = align256(off + sizeof(float) * NN * DD);
    float*     xcur = (float*)(ws + off);     off = align256(off + sizeof(float) * NN * DD);
    _Float16*  xh   = (_Float16*)(ws + off);  off = align256(off + sizeof(_Float16) * NN * DD);
    _Float16*  Wt   = (_Float16*)(ws + off);  off = align256(off + sizeof(_Float16) * 4 * DD * DD);

    const int T = 256;
    const int nd = NN * DD;

    // degrees -> deg_inv_sqrt (in place in dis)
    k_deg_init  <<<(NN + T - 1) / T, T, 0, stream>>>(dis);
    k_deg_count <<<(EE + T - 1) / T, T, 0, stream>>>(dst, dis);
    k_deg_rsqrt <<<(NN + T - 1) / T, T, 0, stream>>>(dis);

    // weight prep (f16, transposed)
    k_wprep<<<(4 * DD * DD + T - 1) / T, T, 0, stream>>>(W1, W2, W3, W4, Wt);

    // input features -> f16 once
    k_tohalf<<<(nd + T - 1) / T, T, 0, stream>>>(x, xh, nd);

    for (int l = 0; l < 4; ++l) {
        // GEMM (WMMA) with fused agg = h * dis^2 epilogue
        k_gemm_wmma<<<NN / 16, 256, 0, stream>>>(xh, Wt + (size_t)l * DD * DD, dis, h, agg);
        // edge aggregation (atomics, L2-resident)
        long long sthreads = (long long)EE * 32;
        k_scatter<<<(unsigned)((sthreads + T - 1) / T), T, 0, stream>>>(src, dst, dis, h, agg);
        // fused ReLU + conversion for next layer's A operand
        k_relu_tohalf<<<(nd + T - 1) / T, T, 0, stream>>>(agg, xcur, xh, nd);
    }

    // global_add_pool into xr (zero first: atomics accumulate and graph replays)
    k_zero<<<(GG * DD + T - 1) / T, T, 0, stream>>>(xr_out, GG * DD);
    k_pool<<<(nd + T - 1) / T, T, 0, stream>>>(xcur, bat, xr_out);

    // classifier + log_softmax
    k_classify<<<(NN + T - 1) / T, T, 0, stream>>>(xcur, fcw, fcb, out);
}
